// Network_70111046140444
// MI455X (gfx1250) — matrix operand order repaired and checked by reading, not compiled, not run on hardware
//
#include <hip/hip_runtime.h>

// Problem constants (match reference)
constexpr int ND = 2;      // layers
constexpr int NE = 16;     // experts
constexpr int NH = 1024;   // width
constexpr int NB = 8192;   // tokens
// top-k = 2 hardcoded

// GEMM tiling: 128x128 tile, 8 waves, each wave owns a 2x4 grid of 16x16 C tiles
constexpr int TM = 128;    // gathered tokens per block
constexpr int TN = 128;    // outputs per block
constexpr int KC = 64;     // k-chunk
constexpr int LDA = 72;    // padded LDS stride (halves): 144B rows -> conflict-friendly

typedef __attribute__((ext_vector_type(16))) _Float16 v16h;
typedef __attribute__((ext_vector_type(8)))  _Float16 v8h;
typedef __attribute__((ext_vector_type(8)))  float    v8f;

__device__ inline v16h pack16(v8h lo, v8h hi) {
  v16h r;
#pragma unroll
  for (int i = 0; i < 8; ++i) { r[i] = lo[i]; r[i + 8] = hi[i]; }
  return r;
}

// ---------------------------------------------------------------------------
// Kernel 1: gating + top-2 routing.
// Block = 128 threads, handles 8 tokens. Per-(token,expert) dot product from
// an LDS-staged X tile, then top-2 + softmax + atomic append to expert lists.
// ---------------------------------------------------------------------------
__global__ __launch_bounds__(128)
void gate_route(const float* __restrict__ x,
                const float* __restrict__ gate_w,
                const float* __restrict__ gate_b,
                int d,
                int*   __restrict__ counts,
                int*   __restrict__ toklist,
                float* __restrict__ gwlist)
{
  __shared__ __align__(16) float Xs[8][NH];
  __shared__ float lg[8][NE];

  const int tid   = threadIdx.x;
  const int tbase = blockIdx.x * 8;

  // Stage 8 token rows: 16 threads per row, float4 loads.
  {
    const int row = tid >> 4;
    const int lc  = tid & 15;
    const float4* src = (const float4*)(x + (size_t)(tbase + row) * NH);
#pragma unroll
    for (int j = 0; j < NH / 64; ++j) {         // 16 iters of float4
      int c4 = j * 16 + lc;
      float4 v = src[c4];
      *(float4*)&Xs[row][c4 * 4] = v;
    }
  }
  __syncthreads();

  // One thread per (token, expert) pair.
  {
    const int t = tid >> 4;
    const int e = tid & 15;
    const float4* wrow = (const float4*)(gate_w + ((size_t)d * NE + e) * NH);
    const float4* xr   = (const float4*)&Xs[t][0];
    float acc = 0.f;
#pragma unroll 4
    for (int i = 0; i < NH / 4; ++i) {
      float4 a = xr[i], b = wrow[i];
      acc += a.x * b.x + a.y * b.y + a.z * b.z + a.w * b.w;
    }
    lg[t][e] = acc + gate_b[d * NE + e];
  }
  __syncthreads();

  // Top-2 + softmax + append (1 thread per token).
  if (tid < 8) {
    float v0 = -1e30f, v1 = -1e30f;
    int   i0 = 0, i1 = 0;
    for (int j = 0; j < NE; ++j) {
      float v = lg[tid][j];
      if (v > v0)      { v1 = v0; i1 = i0; v0 = v; i0 = j; }
      else if (v > v1) { v1 = v;  i1 = j; }
    }
    float e1 = __expf(v1 - v0);
    float w0 = 1.f / (1.f + e1);
    float w1 = e1 / (1.f + e1);
    int token = tbase + tid;
    int s0 = atomicAdd(&counts[i0], 1);
    toklist[i0 * NB + s0] = token; gwlist[i0 * NB + s0] = w0;
    int s1 = atomicAdd(&counts[i1], 1);
    toklist[i1 * NB + s1] = token; gwlist[i1 * NB + s1] = w1;
  }
}

// ---------------------------------------------------------------------------
// Kernel 2: grouped expert GEMM (gather -> f16 WMMA -> PReLU -> gated scatter).
// grid = (NH/TN, NB/TM, NE); block = 256 threads = 8 waves.
// Wave w owns C tiles: rows {2*(w&3), +1} x cols {4*(w>>2) .. +3}.
// Out-of-range gathered rows are CLAMPED (not zeroed): they compute garbage
// rows that are never scattered (epilogue guards gm < cnt), which keeps the
// staging loop divergence-free.
// ---------------------------------------------------------------------------
__global__ __launch_bounds__(256)
void expert_gemm(const float* __restrict__ x,
                 const float* __restrict__ ew,
                 const float* __restrict__ eb,
                 const float* __restrict__ pl,
                 const float* __restrict__ pr,
                 const float* __restrict__ pp,
                 const float* __restrict__ pb,
                 int d,
                 const int*   __restrict__ counts,
                 const int*   __restrict__ toklist,
                 const float* __restrict__ gwlist,
                 float* __restrict__ out)
{
  const int e     = blockIdx.z;
  const int cnt   = counts[e];
  const int mbase = blockIdx.y * TM;
  if (mbase >= cnt) return;                    // uniform early-exit
  const int nbase = blockIdx.x * TN;

  __shared__ __align__(16) _Float16 As[TM * LDA];
  __shared__ __align__(16) _Float16 Ws[TN * LDA];

  const int tid  = threadIdx.x;
  const int wave = tid >> 5;
  const int lane = tid & 31;

  // Staging map: 2 threads per row, 32 halves each.
  const int srow = tid >> 1;                   // 0..127
  const int scol = (tid & 1) * 32;             // 0 or 32

  // Clamped gather: always a valid address, no divergence.
  const int aslot  = min(mbase + srow, cnt - 1);
  const int atoken = toklist[e * NB + aslot];
  const float* asrc = x + (size_t)atoken * NH;
  const float* wsrc = ew + (((size_t)d * NE + e) * NH + (nbase + srow)) * (size_t)NH;

  v8f acc[2][4] = {};
  const int rt0     = (wave & 3) * 2;          // first of 2 row tiles
  const int ct0     = (wave >> 2) * 4;         // first of 4 col tiles
  const int halfsel = lane >> 4;               // 0: lanes 0-15, 1: lanes 16-31
  const int lrow    = lane & 15;

  for (int k0 = 0; k0 < NH; k0 += KC) {
    // --- stage A (gathered tokens) and W (expert weights), f32 -> f16 ---
#pragma unroll
    for (int j = 0; j < 32; j += 4) {
      float4 v = *(const float4*)(asrc + k0 + scol + j);
      _Float16* dp = &As[srow * LDA + scol + j];
      dp[0] = (_Float16)v.x; dp[1] = (_Float16)v.y;
      dp[2] = (_Float16)v.z; dp[3] = (_Float16)v.w;
    }
#pragma unroll
    for (int j = 0; j < 32; j += 4) {
      float4 v = *(const float4*)(wsrc + k0 + scol + j);
      _Float16* dp = &Ws[srow * LDA + scol + j];
      dp[0] = (_Float16)v.x; dp[1] = (_Float16)v.y;
      dp[2] = (_Float16)v.z; dp[3] = (_Float16)v.w;
    }
    if (k0 + KC < NH) {                        // hint next chunks
      __builtin_prefetch(wsrc + k0 + KC + scol, 0, 1);
      __builtin_prefetch(asrc + k0 + KC + scol, 0, 1);
    }
    __syncthreads();

    // --- 2 k-steps of 32: 2 A-frags + 4 B-frags -> 8 WMMAs per wave ---
#pragma unroll
    for (int kk = 0; kk < KC; kk += 32) {
      const int aoff = halfsel * 8;            // ISA A layout: K {0..7,16..23}/{8..15,24..31}
      v16h a[2];
#pragma unroll
      for (int r = 0; r < 2; ++r) {
        int row = (rt0 + r) * 16 + lrow;
        v8h lo = *(const v8h*)&As[row * LDA + kk + aoff];
        v8h hi = *(const v8h*)&As[row * LDA + kk + 16 + aoff];
        a[r] = pack16(lo, hi);
      }
      const int koff = halfsel * 8;            // B layout matches A: K {0..7,16..23}/{8..15,24..31}
#pragma unroll
      for (int c = 0; c < 4; ++c) {
        int col = (ct0 + c) * 16 + lrow;
        v8h lo = *(const v8h*)&Ws[col * LDA + kk + koff];
        v8h hi = *(const v8h*)&Ws[col * LDA + kk + 16 + koff];
        v16h b = pack16(lo, hi);
        acc[0][c] = __builtin_amdgcn_wmma_f32_16x16x32_f16(false, a[0], false, b,
                                                           (short)0, acc[0][c], false, false);
        acc[1][c] = __builtin_amdgcn_wmma_f32_16x16x32_f16(false, a[1], false, b,
                                                           (short)0, acc[1][c], false, false);
      }
    }
    __syncthreads();
  }

  // --- epilogue: bias + custom PReLU + gate-weighted scatter-add ---
  const size_t pbase = ((size_t)d * NE + e) * NH;
#pragma unroll
  for (int r = 0; r < 2; ++r) {
#pragma unroll
    for (int vr = 0; vr < 8; ++vr) {
      int mloc = (rt0 + r) * 16 + vr + halfsel * 8;  // ISA C layout: VGPR r -> row r / r+8
      int gm   = mbase + mloc;
      if (gm >= cnt) continue;
      int   token = toklist[e * NB + gm];
      float wgt   = gwlist[e * NB + gm];
#pragma unroll
      for (int c = 0; c < 4; ++c) {
        int o = nbase + (ct0 + c) * 16 + lrow;
        float y  = acc[r][c][vr] + eb[pbase + o];
        float dl = y - pp[pbase + o];
        float av = pb[pbase + o] + (dl >= 0.f ? dl * pr[pbase + o]
                                              : dl * pl[pbase + o]);
        atomicAdd(&out[(size_t)token * NH + o], wgt * av);
      }
    }
  }
}

// ---------------------------------------------------------------------------
// Host launcher
// ---------------------------------------------------------------------------
extern "C" void kernel_launch(void* const* d_in, const int* in_sizes, int n_in,
                              void* d_out, int out_size, void* d_ws, size_t ws_size,
                              hipStream_t stream) {
  const float* X      = (const float*)d_in[0];
  const float* gate_w = (const float*)d_in[1];
  const float* gate_b = (const float*)d_in[2];
  const float* ew     = (const float*)d_in[3];
  const float* eb     = (const float*)d_in[4];
  const float* pl     = (const float*)d_in[5];
  const float* pr     = (const float*)d_in[6];
  const float* pp     = (const float*)d_in[7];
  const float* pb     = (const float*)d_in[8];
  float* out = (float*)d_out;

  // Workspace layout
  char*  ws      = (char*)d_ws;
  int*   counts  = (int*)ws;                                    // 256 B slot
  int*   toklist = (int*)(ws + 256);                            // NE*NB*4
  float* gwlist  = (float*)(ws + 256 + (size_t)NE * NB * 4);    // NE*NB*4
  float* x1      = (float*)(ws + 256 + (size_t)NE * NB * 8);    // NB*NH*4

  for (int d = 0; d < ND; ++d) {
    const float* xin  = (d == 0) ? X : x1;
    float*       xout = (d == ND - 1) ? out : x1;

    hipMemsetAsync(counts, 0, NE * sizeof(int), stream);
    hipMemsetAsync(xout, 0, (size_t)NB * NH * sizeof(float), stream);

    gate_route<<<NB / 8, 128, 0, stream>>>(xin, gate_w, gate_b, d,
                                           counts, toklist, gwlist);

    dim3 grid(NH / TN, NB / TM, NE);
    expert_gemm<<<grid, 256, 0, stream>>>(xin, ew, eb, pl, pr, pp, pb, d,
                                          counts, toklist, gwlist, xout);
  }
}
